// SpatialBranch_31739808317486
// MI455X (gfx1250) — matrix operand order repaired and checked by reading, not compiled, not run on hardware
//
#include <hip/hip_runtime.h>
#include <hip/hip_bf16.h>

// ---------------------------------------------------------------------------
// GCN (6 layers, dims 128-32-16-8-16-32-128) on N=100000 (= 6250*16) nodes,
// E=1.6M edges. Edge scatter is the bandwidth bound (~3 GB total, mostly
// L2-resident); GEMMs run on v_wmma_f32_16x16x32_f16 with B pre-packed into
// the WMMA register layout and BN+ReLU fused into the A-operand conversion.
// ---------------------------------------------------------------------------

typedef __attribute__((ext_vector_type(16))) _Float16 v16h;
typedef __attribute__((ext_vector_type(8)))  float    v8f;

#define WMMA_EPS 1e-5f

// ---------------- small utility kernels ----------------
__global__ void k_zero(float* __restrict__ p, long long n) {
    long long i = (long long)blockIdx.x * blockDim.x + threadIdx.x;
    if (i < n) p[i] = 0.f;
}

// deg[col[e]] += ew[e]
__global__ void k_deg(const long long* __restrict__ ei, const float* __restrict__ ew,
                      float* __restrict__ deg, long long E) {
    long long e = (long long)blockIdx.x * blockDim.x + threadIdx.x;
    if (e < E) unsafeAtomicAdd(&deg[ei[E + e]], ew[e]);
}

// dinv = rsqrt(deg+1); self_norm = dinv*dinv
__global__ void k_dinv(const float* __restrict__ deg, float* __restrict__ dinv,
                       float* __restrict__ self_norm, int n) {
    int i = blockIdx.x * blockDim.x + threadIdx.x;
    if (i < n) {
        float di = rsqrtf(deg[i] + 1.0f);
        dinv[i] = di;
        self_norm[i] = di * di;
    }
}

// norm_e[e] = dinv[row]*ew*dinv[col]
__global__ void k_norm_e(const long long* __restrict__ ei, const float* __restrict__ ew,
                         const float* __restrict__ dinv, float* __restrict__ norm_e,
                         long long E) {
    long long e = (long long)blockIdx.x * blockDim.x + threadIdx.x;
    if (e < E) norm_e[e] = dinv[ei[e]] * ew[e] * dinv[ei[E + e]];
}

// ---------------- B pre-pack: W[F_in,F_out] f32 -> WMMA B layout f16 --------
// Bp[(kt*numCt + ct)*32 + lane] = v16h with halfs j: K = kt*32 + (lane>>4)*8 + (j&7) + (j>>3)*16,
// N = ct*16 + (lane&15); zero-padded outside [F_in, F_out].
__global__ void k_pack_B(const float* __restrict__ W, v16h* __restrict__ Bp,
                         int F_in, int F_out, int numCt) {
    const int lane = threadIdx.x;          // one wave per block
    const int tile = blockIdx.x;           // kt * numCt + ct
    const int kt = tile / numCt, ct = tile % numCt;
    const int half_id = lane >> 4, lane16 = lane & 15;
    const int coln = ct * 16 + lane16;
    v16h b;
#pragma unroll
    for (int j = 0; j < 16; ++j) {
        int kk = kt * 32 + half_id * 8 + (j & 7) + ((j >> 3) * 16);
        float v = (kk < F_in && coln < F_out) ? W[(long long)kk * F_out + coln] : 0.f;
        b[j] = (_Float16)v;
    }
    Bp[(long long)tile * 32 + lane] = b;
}

// ---------------- WMMA GEMM + fused epilogue --------------------------------
// t[N,F_OUT]   = act(in)[N,F_IN] @ W          (pre-aggregation h, for gather)
// out[N,F_OUT] = self_norm[:,None]*t + bias   (aggregation seed; atomics add on top)
// act = optional BN affine + ReLU on the input features (fused into f32->f16 cvt).
template <int F_IN, int F_OUT, bool BN, bool RELU>
__global__ void k_gemm(const float* __restrict__ in, const v16h* __restrict__ Bp,
                       const float* __restrict__ bn_scale, const float* __restrict__ bn_shift,
                       const float* __restrict__ self_norm, const float* __restrict__ bias,
                       float* __restrict__ t, float* __restrict__ out, int ntiles) {
    constexpr int KT  = (F_IN + 31) / 32;    // 32-wide K tiles
    constexpr int NCT = (F_OUT + 15) / 16;   // 16-wide N tiles
    const int lane    = threadIdx.x & 31;
    const int wave    = threadIdx.x >> 5;    // 4 waves / block
    const int half_id = lane >> 4;
    const int lane16  = lane & 15;
    const int ct      = blockIdx.y;
    const int coln    = ct * 16 + lane16;

    // B for this column tile: held in registers across all row tiles.
    v16h breg[KT];
#pragma unroll
    for (int kt = 0; kt < KT; ++kt)
        breg[kt] = Bp[(long long)(kt * NCT + ct) * 32 + lane];

    // Per-lane BN scale/shift for the 16 K indices this lane feeds to WMMA.
    float sc[16], sh[16];
    if constexpr (BN) {
#pragma unroll
        for (int j = 0; j < 16; ++j) {       // KT==1 whenever BN is on (F_IN<=32)
            int kk = half_id * 8 + (j & 7) + ((j >> 3) * 16);
            bool ok = kk < F_IN;
            sc[j] = ok ? bn_scale[kk] : 0.f;
            sh[j] = ok ? bn_shift[kk] : 0.f;
        }
    }

    float biasv;
    if constexpr (F_OUT % 16 == 0) biasv = bias[coln];
    else                           biasv = (lane16 < F_OUT) ? bias[coln] : 0.f;

    for (int tile = blockIdx.x * 4 + wave; tile < ntiles; tile += gridDim.x * 4) {
        const int row = tile * 16 + lane16;
        const float4* rowp = (const float4*)(in + (long long)row * F_IN);

        v8f c = {};
#pragma unroll
        for (int kt = 0; kt < KT; ++kt) {
            const int k0 = kt * 32;
            // Two contiguous 8-float runs per lane (16-bit A layout):
            //   run1: K = k0 + half_id*8 + {0..7},  run2: K = k0+16 + half_id*8 + {0..7}
            float4 p0 = {0, 0, 0, 0}, p1 = {0, 0, 0, 0}, p2 = {0, 0, 0, 0}, p3 = {0, 0, 0, 0};
            if constexpr (F_IN % 32 == 0) {
                int i0 = (k0 + half_id * 8) >> 2;
                p0 = rowp[i0]; p1 = rowp[i0 + 1];
                int i1 = (k0 + 16 + half_id * 8) >> 2;
                p2 = rowp[i1]; p3 = rowp[i1 + 1];
            } else if constexpr (F_IN == 16) {
                int i0 = (half_id * 8) >> 2;              // run2 entirely K>=16: zero
                p0 = rowp[i0]; p1 = rowp[i0 + 1];
            } else {                                       // F_IN == 8
                if (half_id == 0) { p0 = rowp[0]; p1 = rowp[1]; }
            }
            float av[16] = {p0.x, p0.y, p0.z, p0.w, p1.x, p1.y, p1.z, p1.w,
                            p2.x, p2.y, p2.z, p2.w, p3.x, p3.y, p3.z, p3.w};
            v16h a;
#pragma unroll
            for (int j = 0; j < 16; ++j) {
                float v = av[j];
                if constexpr (BN)   v = fmaf(v, sc[j], sh[j]);
                if constexpr (RELU) v = v > 0.f ? v : 0.f;
                a[j] = (_Float16)v;
            }
            c = __builtin_amdgcn_wmma_f32_16x16x32_f16(
                    false, a, false, breg[kt], (short)0, c, false, false);
        }

        // Epilogue: D rows for this lane are M = half_id*8 + {0..7} (contiguous).
        const int rbase = tile * 16 + half_id * 8;
        float4 sn0 = *(const float4*)(self_norm + rbase);
        float4 sn1 = *(const float4*)(self_norm + rbase + 4);
        float snv[8] = {sn0.x, sn0.y, sn0.z, sn0.w, sn1.x, sn1.y, sn1.z, sn1.w};
#pragma unroll
        for (int r = 0; r < 8; ++r) {
            long long o = (long long)(rbase + r) * F_OUT + coln;
            if constexpr (F_OUT % 16 == 0) {
                t[o]   = c[r];
                out[o] = snv[r] * c[r] + biasv;
            } else {
                if (lane16 < F_OUT) { t[o] = c[r]; out[o] = snv[r] * c[r] + biasv; }
            }
        }
    }
}

// out[col[e],f] += norm_e[e] * h[row[e],f]
__global__ void k_edge_scatter(const float* __restrict__ h, const long long* __restrict__ ei,
                               const float* __restrict__ norm_e, float* __restrict__ out,
                               int logF, long long E) {
    long long idx = (long long)blockIdx.x * blockDim.x + threadIdx.x;
    long long e = idx >> logF;
    if (e < E) {
        int f = (int)(idx & ((1 << logF) - 1));
        long long r = ei[e];
        long long c = ei[E + e];
        float v = norm_e[e] * h[(r << logF) + f];
        unsafeAtomicAdd(&out[(c << logF) + f], v);
    }
}

// per-feature sum / sumsq over N rows (F = power of 2, <= 32)
__global__ void k_bn_stats(const float* __restrict__ h, int n, int F,
                           float* __restrict__ sum, float* __restrict__ sumsq) {
    __shared__ float s1[256];
    __shared__ float s2[256];
    int tid = threadIdx.x;
    int f = tid & (F - 1);
    int rowInBlock = tid / F;
    int rowsPerBlock = 256 / F;
    float ls = 0.f, lq = 0.f;
    for (long long r = (long long)blockIdx.x * rowsPerBlock + rowInBlock; r < n;
         r += (long long)gridDim.x * rowsPerBlock) {
        float v = h[r * F + f];
        ls += v; lq += v * v;
    }
    s1[tid] = ls; s2[tid] = lq;
    __syncthreads();
    for (int s = 128; s >= F; s >>= 1) {   // F | s, so features stay aligned
        if (tid < s) { s1[tid] += s1[tid + s]; s2[tid] += s2[tid + s]; }
        __syncthreads();
    }
    if (tid < F) {
        unsafeAtomicAdd(&sum[tid],   s1[tid]);
        unsafeAtomicAdd(&sumsq[tid], s2[tid]);
    }
}

// scale = g*rsqrt(var+eps); shift = beta - mean*scale  (consumed by next GEMM)
__global__ void k_bn_finalize(const float* __restrict__ sum, const float* __restrict__ sumsq,
                              const float* __restrict__ g, const float* __restrict__ beta,
                              int n, float* __restrict__ scale, float* __restrict__ shift,
                              int F) {
    int i = threadIdx.x;
    if (i < F) {
        float inv_n = 1.0f / (float)n;
        float mean = sum[i] * inv_n;
        float var  = sumsq[i] * inv_n - mean * mean;
        float s    = g[i] * rsqrtf(var + WMMA_EPS);
        scale[i] = s;
        shift[i] = beta[i] - mean * s;
    }
}

// ---------------------------------------------------------------------------
static inline long long cdivll(long long a, long long b) { return (a + b - 1) / b; }

extern "C" void kernel_launch(void* const* d_in, const int* in_sizes, int n_in,
                              void* d_out, int out_size, void* d_ws, size_t ws_size,
                              hipStream_t stream) {
    const int DIMS[7] = {128, 32, 16, 8, 16, 32, 128};
    const int LOGF[6] = {5, 4, 3, 4, 5, 7};          // log2(F_out) per layer

    const int N = in_sizes[0] / 128;                 // 100000, multiple of 16
    const long long E = in_sizes[2];
    const int ntiles = N / 16;                       // 6250

    const float*     x  = (const float*)d_in[0];
    const long long* ei = (const long long*)d_in[1]; // int64 (2,E)
    const float*     ew = (const float*)d_in[2];
    const float* Wm[6]; const float* bm[6];
    for (int i = 0; i < 6; ++i) { Wm[i] = (const float*)d_in[3 + 2 * i];  bm[i] = (const float*)d_in[4 + 2 * i]; }
    const float* gm[5]; const float* betam[5];
    for (int i = 0; i < 5; ++i) { gm[i] = (const float*)d_in[15 + 2 * i]; betam[i] = (const float*)d_in[16 + 2 * i]; }

    // workspace carve-up (floats; every block is a multiple of 32 bytes)
    float* ws = (float*)d_ws;
    float* deg       = ws;                            // N
    float* dinv      = deg + N;                       // N
    float* self_norm = dinv + N;                      // N
    float* norm_e    = self_norm + N;                 // E
    float* t_buf     = norm_e + E;                    // N*128
    float* o_a       = t_buf + (long long)N * 128;    // N*32
    float* o_b       = o_a + (long long)N * 32;       // N*32
    float* bn_sum    = o_b + (long long)N * 32;       // 32
    float* bn_sumsq  = bn_sum + 32;                   // 32
    float* bn_scale  = bn_sumsq + 32;                 // 32
    float* bn_shift  = bn_scale + 32;                 // 32
    v16h*  Bp        = (v16h*)(bn_shift + 32);        // 4K halfs max (8KB)

    // ---- gcn_norm ----
    k_zero<<<cdivll(N, 256), 256, 0, stream>>>(deg, N);
    k_deg<<<cdivll(E, 256), 256, 0, stream>>>(ei, ew, deg, E);
    k_dinv<<<cdivll(N, 256), 256, 0, stream>>>(deg, dinv, self_norm, N);
    k_norm_e<<<cdivll(E, 256), 256, 0, stream>>>(ei, ew, dinv, norm_e, E);

    // ---- 6 GCN layers ----
    const float* in_ptr = x;
    float* o_bufs[2] = {o_a, o_b};
    for (int i = 0; i < 6; ++i) {
        const int F_in = DIMS[i], F_out = DIMS[i + 1], logF = LOGF[i];
        const int KT = (F_in + 31) / 32, NCT = (F_out + 15) / 16;
        float* outb = (i == 5) ? (float*)d_out : o_bufs[i & 1];

        k_pack_B<<<KT * NCT, 32, 0, stream>>>(Wm[i], Bp, F_in, F_out, NCT);

        dim3 ggrid(160, NCT);                         // waves grid-stride row tiles
        switch (i) {  // <F_IN, F_OUT, BN, RELU-on-input>
        case 0: k_gemm<128,  32, false, false><<<ggrid, 128, 0, stream>>>(
                    in_ptr, Bp, nullptr, nullptr, self_norm, bm[i], t_buf, outb, ntiles); break;
        case 1: k_gemm< 32,  16, true,  true ><<<ggrid, 128, 0, stream>>>(
                    in_ptr, Bp, bn_scale, bn_shift, self_norm, bm[i], t_buf, outb, ntiles); break;
        case 2: k_gemm< 16,   8, true,  true ><<<ggrid, 128, 0, stream>>>(
                    in_ptr, Bp, bn_scale, bn_shift, self_norm, bm[i], t_buf, outb, ntiles); break;
        case 3: k_gemm<  8,  16, true,  false><<<ggrid, 128, 0, stream>>>(
                    in_ptr, Bp, bn_scale, bn_shift, self_norm, bm[i], t_buf, outb, ntiles); break;
        case 4: k_gemm< 16,  32, true,  true ><<<ggrid, 128, 0, stream>>>(
                    in_ptr, Bp, bn_scale, bn_shift, self_norm, bm[i], t_buf, outb, ntiles); break;
        case 5: k_gemm< 32, 128, true,  true ><<<ggrid, 128, 0, stream>>>(
                    in_ptr, Bp, bn_scale, bn_shift, self_norm, bm[i], t_buf, outb, ntiles); break;
        }

        long long totalEF = E << logF;
        k_edge_scatter<<<cdivll(totalEF, 256), 256, 0, stream>>>(t_buf, ei, norm_e,
                                                                 outb, logF, E);
        if (i < 5) {
            k_zero<<<1, 64, 0, stream>>>(bn_sum, 64);   // bn_sum + bn_sumsq contiguous
            k_bn_stats<<<512, 256, 0, stream>>>(outb, N, F_out, bn_sum, bn_sumsq);
            k_bn_finalize<<<1, F_out, 0, stream>>>(bn_sum, bn_sumsq, gm[i], betam[i],
                                                   N, bn_scale, bn_shift, F_out);
        }
        in_ptr = outb;
    }
}
